// ablation3_51213190038001
// MI455X (gfx1250) — matrix operand order repaired and checked by reading, not compiled, not run on hardware
//
#include <hip/hip_runtime.h>

// ---------------------------------------------------------------------------
// CDNA5 (gfx1250) implementation of the 4-head GAT reference.
// GEMMs: v_wmma_f32_16x16x32_f16, double-buffered LDS tiles filled by
// global_load_async_to_lds_b128 (ASYNCcnt-tracked), one barrier per K-step.
// All B operands are pre-transposed (NxK) so staging is pure 16B async loads.
// B fragments are bulk-loaded into registers before the WMMA burst so the
// scheduler can overlap ds_load latency against matrix-pipe issue.
// ---------------------------------------------------------------------------

typedef __attribute__((ext_vector_type(16))) _Float16 v16h;
typedef __attribute__((ext_vector_type(8)))  _Float16 v8h;
typedef __attribute__((ext_vector_type(8)))  float    v8f;

#define S_CNT 16384
#define R_CNT 1024
#define F_IN  512
#define F_OUT 256
#define NH    4
#define ALPHA 0.2f
#define NEGV  -9.0e15f
#define BN_EPS 1e-5f

__device__ __forceinline__ float lrelu(float x){ return x > 0.f ? x : ALPHA * x; }
__device__ __forceinline__ float eluf (float x){ return x > 0.f ? x : (__expf(x) - 1.f); }

// Async global->LDS 16B copy (saddr form: 32-bit voffset + uniform 64-bit base).
__device__ __forceinline__ void async_b128(uint32_t lds_addr, uint32_t voff, uint64_t sbase)
{
  asm volatile("global_load_async_to_lds_b128 %0, %1, %2"
               :: "v"(lds_addr), "v"(voff), "s"(sbase) : "memory");
}
__device__ __forceinline__ void wait_async0()
{
  asm volatile("s_wait_asynccnt 0" ::: "memory");
}

// ---------------------------------------------------------------------------
// WMMA GEMM: D = A(f16, MxK row-major) * B^T(f16, NxK row-major)
//   EPI=0: store f32 into D (ldd).  EPI=1: elu -> f16 into Dh (lddh).
// Block = 256 threads = 8 waves; tile 128x128x32; wave w owns rows [16w,16w+16).
// M%128==0, N%128==0, K%32==0 (true for every GEMM here).
// ---------------------------------------------------------------------------
template<int EPI>
__global__ __launch_bounds__(256)
void gemm_wmma(const _Float16* __restrict__ A, int lda,
               const _Float16* __restrict__ B, int ldb,
               float* __restrict__ D, int ldd,
               _Float16* __restrict__ Dh, int lddh,
               int M, int N, int K)
{
  constexpr int BK   = 32;
  constexpr int LDSK = BK + 8;                 // +16B pad against bank conflicts
  __shared__ _Float16 As[2][128 * LDSK];
  __shared__ _Float16 Bs[2][128 * LDSK];       // stored [n][k]

  const int bm   = blockIdx.y * 128;
  const int bn   = blockIdx.x * 128;
  const int tid  = threadIdx.x;
  const int wave = tid >> 5;
  const int lane = tid & 31;
  const int l16  = lane & 15;
  const int hi   = lane >> 4;

  // this thread's two staging chunks: (row, 8-half column offset)
  const int r0 = tid >> 2;                     // 0..63
  const int c0 = (tid & 3) * 8;                // 0,8,16,24
  const int r1 = r0 + 64;                      // 64..127

  const uint64_t Abase = (uint64_t)(uintptr_t)A;
  const uint64_t Bbase = (uint64_t)(uintptr_t)B;

  v8f acc[8];
  #pragma unroll
  for (int t = 0; t < 8; ++t)
    #pragma unroll
    for (int e = 0; e < 8; ++e) acc[t][e] = 0.f;

  const int nk = K / BK;

  auto stage = [&](int b, int k0) {
    uint32_t ao0 = (uint32_t)((bm + r0) * lda + k0 + c0) * 2u;
    uint32_t ao1 = (uint32_t)((bm + r1) * lda + k0 + c0) * 2u;
    uint32_t bo0 = (uint32_t)((bn + r0) * ldb + k0 + c0) * 2u;
    uint32_t bo1 = (uint32_t)((bn + r1) * ldb + k0 + c0) * 2u;
    async_b128((uint32_t)(uintptr_t)&As[b][r0 * LDSK + c0], ao0, Abase);
    async_b128((uint32_t)(uintptr_t)&As[b][r1 * LDSK + c0], ao1, Abase);
    async_b128((uint32_t)(uintptr_t)&Bs[b][r0 * LDSK + c0], bo0, Bbase);
    async_b128((uint32_t)(uintptr_t)&Bs[b][r1 * LDSK + c0], bo1, Bbase);
    if (k0 + 2 * BK < K) {                     // L2 prefetch two tiles ahead
      __builtin_prefetch(A + (size_t)(bm + r0) * lda + k0 + 2 * BK, 0, 1);
      __builtin_prefetch(B + (size_t)(bn + r0) * ldb + k0 + 2 * BK, 0, 1);
    }
  };

  stage(0, 0);
  wait_async0();
  __syncthreads();

  int buf = 0;
  for (int t = 0; t < nk; ++t) {
    if (t + 1 < nk) stage(buf ^ 1, (t + 1) * BK);

    // A fragment (16x32 f16: lanes<16 hold K {0..7,16..23}, lanes>=16 {8..15,24..31})
    const int arow = wave * 16 + l16;
    const int akb  = hi ? 8 : 0;
    v8h alo = *(const v8h*)(&As[buf][arow * LDSK + akb]);
    v8h ahi = *(const v8h*)(&As[buf][arow * LDSK + akb + 16]);
    v16h af;
    #pragma unroll
    for (int e = 0; e < 8; ++e) { af[e] = alo[e]; af[e + 8] = ahi[e]; }

    // Bulk-load all 8 B fragments (32x16: lanes<16 K {0..7,16..23}, lanes>=16 {8..15,24..31})
    // into distinct registers first -> 16 ds_load_b128 in flight, dscnt counts
    // down across the WMMA burst instead of stalling to 0 per WMMA.
    v8h bfr0[8], bfr1[8];
    #pragma unroll
    for (int nt = 0; nt < 8; ++nt) {
      const int bcol = nt * 16 + l16;
      const int bkb  = hi ? 8 : 0;
      bfr0[nt] = *(const v8h*)(&Bs[buf][bcol * LDSK + bkb]);
      bfr1[nt] = *(const v8h*)(&Bs[buf][bcol * LDSK + bkb + 16]);
    }
    #pragma unroll
    for (int nt = 0; nt < 8; ++nt) {
      v16h bf;
      #pragma unroll
      for (int e = 0; e < 8; ++e) { bf[e] = bfr0[nt][e]; bf[e + 8] = bfr1[nt][e]; }
      acc[nt] = __builtin_amdgcn_wmma_f32_16x16x32_f16(
          false, af, false, bf, (short)0, acc[nt], false, false);
    }

    wait_async0();       // next tile fully landed in LDS (this wave's asyncs)
    __syncthreads();     // all waves done reading buf + done waiting
    buf ^= 1;
  }

  // epilogue: C layout (VGPR r: lanes<16 -> M=r, lanes>=16 -> M=r+8; N=lane&15)
  #pragma unroll
  for (int r = 0; r < 8; ++r) {
    const int row = bm + wave * 16 + r + hi * 8;
    if constexpr (EPI == 0) {
      float* p = D + (size_t)row * ldd + bn + l16;
      #pragma unroll
      for (int nt = 0; nt < 8; ++nt) p[nt * 16] = acc[nt][r];
    } else {
      _Float16* p = Dh + (size_t)row * lddh + bn + l16;
      #pragma unroll
      for (int nt = 0; nt < 8; ++nt) p[nt * 16] = (_Float16)eluf(acc[nt][r]);
    }
  }
}

// ---------------------------------------------------------------------------
// Elementwise / reduction helpers
// ---------------------------------------------------------------------------
__global__ void cvt_f16_kernel(const float* __restrict__ x, _Float16* __restrict__ y, size_t n)
{
  size_t i = (size_t)blockIdx.x * blockDim.x + threadIdx.x;
  size_t st = (size_t)gridDim.x * blockDim.x;
  for (; i < n; i += st) y[i] = (_Float16)x[i];
}

// W (NH,F_IN,F_OUT) f32 -> WcT ((NH*F_OUT) x F_IN) f16: WcT[h*F_OUT+c][k] = W[h][k][c]
__global__ void cvt_catT_kernel(const float* __restrict__ W, _Float16* __restrict__ WcT)
{
  size_t n = (size_t)NH * F_OUT * F_IN;
  size_t o = (size_t)blockIdx.x * blockDim.x + threadIdx.x;
  size_t st = (size_t)gridDim.x * blockDim.x;
  for (; o < n; o += st) {
    int k = (int)(o & (F_IN - 1));
    int nr = (int)(o >> 9);              // F_IN = 512
    int h = nr >> 8;
    int c = nr & 255;
    WcT[o] = (_Float16)W[((size_t)(h * F_IN + k)) * F_OUT + c];
  }
}

// out[c*rows + r] = (f16) in[r*cols + c]; rows, cols multiples of 32.
__global__ __launch_bounds__(256)
void transpose_cvt_kernel(const float* __restrict__ in, _Float16* __restrict__ out,
                          int rows, int cols)
{
  __shared__ float tile[32][33];
  const int bc = blockIdx.x * 32;
  const int br = blockIdx.y * 32;
  const int tx = threadIdx.x & 31;
  const int ty = threadIdx.x >> 5;       // 0..7
  #pragma unroll
  for (int i = ty; i < 32; i += 8)
    tile[i][tx] = in[(size_t)(br + i) * cols + bc + tx];
  __syncthreads();
  #pragma unroll
  for (int i = ty; i < 32; i += 8)
    out[(size_t)(bc + i) * rows + br + tx] = (_Float16)tile[tx][i];
}

__global__ void av_kernel(const float* __restrict__ a_out, float* __restrict__ av)
{
  int j = blockIdx.x * blockDim.x + threadIdx.x;
  if (j < R_CNT) av[j] = a_out[j] + a_out[R_CNT + j];
}

// out[i] = sum_c X[i, coff+c] * a[c], c < 256. One block (256 thr) per row.
__global__ __launch_bounds__(256)
void dot256_kernel(const float* __restrict__ X, int ldx, int coff,
                   const float* __restrict__ a, float* __restrict__ out)
{
  __shared__ float red[256];
  const int i = blockIdx.x;
  red[threadIdx.x] = X[(size_t)i * ldx + coff + threadIdx.x] * a[threadIdx.x];
  __syncthreads();
  for (int s = 128; s > 0; s >>= 1) {
    if (threadIdx.x < s) red[threadIdx.x] += red[threadIdx.x + s];
    __syncthreads();
  }
  if (threadIdx.x == 0) out[i] = red[0];
}

// Per-row masked softmax of lrelu(s2[i] + r1[j]); writes att (SxR) and att^T (RxS) f16.
__global__ __launch_bounds__(256)
void att_kernel(const float* __restrict__ s2, const float* __restrict__ r1,
                const int* __restrict__ adj,
                _Float16* __restrict__ attF, _Float16* __restrict__ attT)
{
  __shared__ float red[256];
  const int i = blockIdx.x;
  const float si = s2[i];
  float xv[4];
  float m = -3.4e38f;
  #pragma unroll
  for (int t = 0; t < 4; ++t) {
    int j = threadIdx.x + t * 256;
    float e = lrelu(si + r1[j]);
    float x = (adj[(size_t)i * R_CNT + j] > 0) ? e : NEGV;
    xv[t] = x;
    m = fmaxf(m, x);
  }
  red[threadIdx.x] = m; __syncthreads();
  for (int s = 128; s > 0; s >>= 1) {
    if (threadIdx.x < s) red[threadIdx.x] = fmaxf(red[threadIdx.x], red[threadIdx.x + s]);
    __syncthreads();
  }
  m = red[0]; __syncthreads();
  float ps[4], lsum = 0.f;
  #pragma unroll
  for (int t = 0; t < 4; ++t) { ps[t] = __expf(xv[t] - m); lsum += ps[t]; }
  red[threadIdx.x] = lsum; __syncthreads();
  for (int s = 128; s > 0; s >>= 1) {
    if (threadIdx.x < s) red[threadIdx.x] += red[threadIdx.x + s];
    __syncthreads();
  }
  const float inv = 1.f / red[0];
  #pragma unroll
  for (int t = 0; t < 4; ++t) {
    int j = threadIdx.x + t * 256;
    float a = ps[t] * inv;
    attF[(size_t)i * R_CNT + j] = (_Float16)a;
    attT[(size_t)j * S_CNT + i] = (_Float16)a;
  }
}

// Column-wise mean/var (biased) over `rows` rows of X (ld = 256). One block per column.
__global__ __launch_bounds__(256)
void bn_stats_kernel(const float* __restrict__ X, int rows,
                     float* __restrict__ mu, float* __restrict__ var)
{
  __shared__ float rs[256], rq[256];
  const int c = blockIdx.x;
  float s = 0.f, q = 0.f;
  for (int r = threadIdx.x; r < rows; r += 256) {
    float v = X[(size_t)r * F_OUT + c];
    s += v; q += v * v;
  }
  rs[threadIdx.x] = s; rq[threadIdx.x] = q; __syncthreads();
  for (int st = 128; st > 0; st >>= 1) {
    if (threadIdx.x < st) { rs[threadIdx.x] += rs[threadIdx.x + st]; rq[threadIdx.x] += rq[threadIdx.x + st]; }
    __syncthreads();
  }
  if (threadIdx.x == 0) {
    float mean = rs[0] / rows;
    mu[c] = mean;
    var[c] = rq[0] / rows - mean * mean;
  }
}

// y_f16 = lrelu(g*(x-mu)*rsqrt(var+eps)+b); x is rows x 256.
__global__ void bn_apply_kernel(const float* __restrict__ X, size_t n,
                                const float* __restrict__ mu, const float* __restrict__ var,
                                const float* __restrict__ g, const float* __restrict__ b,
                                _Float16* __restrict__ Y)
{
  size_t i = (size_t)blockIdx.x * blockDim.x + threadIdx.x;
  size_t st = (size_t)gridDim.x * blockDim.x;
  for (; i < n; i += st) {
    int c = (int)(i & (F_OUT - 1));
    float v = g[c] * (X[i] - mu[c]) * rsqrtf(var[c] + BN_EPS) + b[c];
    Y[i] = (_Float16)lrelu(v);
  }
}

// e[i] = lrelu(sum_j H[i,j]*av[j]); one block per row.
__global__ __launch_bounds__(256)
void eout_kernel(const float* __restrict__ H, const float* __restrict__ av,
                 float* __restrict__ e)
{
  __shared__ float red[256];
  const int i = blockIdx.x;
  float s = 0.f;
  #pragma unroll
  for (int t = 0; t < 4; ++t) {
    int j = threadIdx.x + t * 256;
    s += H[(size_t)i * R_CNT + j] * av[j];
  }
  red[threadIdx.x] = s; __syncthreads();
  for (int st = 128; st > 0; st >>= 1) {
    if (threadIdx.x < st) red[threadIdx.x] += red[threadIdx.x + st];
    __syncthreads();
  }
  if (threadIdx.x == 0) e[i] = lrelu(red[0]);
}

// Final: per-row masked softmax of broadcast e[i], y = elu(elu(att*h)), log_softmax.
__global__ __launch_bounds__(256)
void final_kernel(const float* __restrict__ H, const float* __restrict__ e,
                  const int* __restrict__ adj, float* __restrict__ out)
{
  __shared__ float red[256];
  const int i = blockIdx.x;
  const float ei = e[i];
  float xv[4];
  float m = -3.4e38f;
  #pragma unroll
  for (int t = 0; t < 4; ++t) {
    int j = threadIdx.x + t * 256;
    float x = (adj[(size_t)i * R_CNT + j] > 0) ? ei : NEGV;
    xv[t] = x;
    m = fmaxf(m, x);
  }
  red[threadIdx.x] = m; __syncthreads();
  for (int s = 128; s > 0; s >>= 1) {
    if (threadIdx.x < s) red[threadIdx.x] = fmaxf(red[threadIdx.x], red[threadIdx.x + s]);
    __syncthreads();
  }
  m = red[0]; __syncthreads();
  float ps[4], lsum = 0.f;
  #pragma unroll
  for (int t = 0; t < 4; ++t) { ps[t] = __expf(xv[t] - m); lsum += ps[t]; }
  red[threadIdx.x] = lsum; __syncthreads();
  for (int s = 128; s > 0; s >>= 1) {
    if (threadIdx.x < s) red[threadIdx.x] += red[threadIdx.x + s];
    __syncthreads();
  }
  const float inv = 1.f / red[0];
  __syncthreads();
  float yv[4];
  float m2 = -3.4e38f;
  #pragma unroll
  for (int t = 0; t < 4; ++t) {
    int j = threadIdx.x + t * 256;
    float att = ps[t] * inv;
    float y = eluf(eluf(att * H[(size_t)i * R_CNT + j]));
    yv[t] = y;
    m2 = fmaxf(m2, y);
  }
  red[threadIdx.x] = m2; __syncthreads();
  for (int s = 128; s > 0; s >>= 1) {
    if (threadIdx.x < s) red[threadIdx.x] = fmaxf(red[threadIdx.x], red[threadIdx.x + s]);
    __syncthreads();
  }
  m2 = red[0]; __syncthreads();
  float es = 0.f;
  #pragma unroll
  for (int t = 0; t < 4; ++t) es += __expf(yv[t] - m2);
  red[threadIdx.x] = es; __syncthreads();
  for (int s = 128; s > 0; s >>= 1) {
    if (threadIdx.x < s) red[threadIdx.x] += red[threadIdx.x + s];
    __syncthreads();
  }
  const float lse = m2 + __logf(red[0]);
  #pragma unroll
  for (int t = 0; t < 4; ++t) {
    int j = threadIdx.x + t * 256;
    out[(size_t)i * R_CNT + j] = yv[t] - lse;
  }
}

// ---------------------------------------------------------------------------
// Host orchestration
// ---------------------------------------------------------------------------
static inline size_t alloc_off(size_t& cur, size_t bytes)
{
  size_t o = cur;
  cur = (cur + bytes + 255) & ~(size_t)255;
  return o;
}

extern "C" void kernel_launch(void* const* d_in, const int* in_sizes, int n_in,
                              void* d_out, int out_size, void* d_ws, size_t ws_size,
                              hipStream_t stream)
{
  (void)in_sizes; (void)n_in; (void)out_size; (void)ws_size;
  const int*   adj   = (const int*)  d_in[0];
  const float* Sfeat = (const float*)d_in[4];
  const float* Rfeat = (const float*)d_in[5];
  const float* W1    = (const float*)d_in[6];
  const float* W2    = (const float*)d_in[7];
  const float* a_att = (const float*)d_in[8];
  const float* bn1_g = (const float*)d_in[9];
  const float* bn1_b = (const float*)d_in[10];
  const float* bn2_g = (const float*)d_in[11];
  const float* bn2_b = (const float*)d_in[12];
  const float* W_out = (const float*)d_in[13];
  const float* a_out = (const float*)d_in[14];
  float* out = (float*)d_out;

  char* ws = (char*)d_ws;
  size_t cur = 0;
  _Float16* Sf16   = (_Float16*)(ws + alloc_off(cur, (size_t)S_CNT * F_IN * 2));
  _Float16* Rf16   = (_Float16*)(ws + alloc_off(cur, (size_t)R_CNT * F_IN * 2));
  _Float16* W1cT   = (_Float16*)(ws + alloc_off(cur, (size_t)NH * F_OUT * F_IN * 2));
  _Float16* W2cT   = (_Float16*)(ws + alloc_off(cur, (size_t)NH * F_OUT * F_IN * 2));
  _Float16* WoT    = (_Float16*)(ws + alloc_off(cur, (size_t)NH * R_CNT * R_CNT * 2));
  float*    h2     = (float*)   (ws + alloc_off(cur, (size_t)S_CNT * NH * F_OUT * 4));
  _Float16* h2T    = (_Float16*)(ws + alloc_off(cur, (size_t)S_CNT * NH * F_OUT * 2));
  float*    h1     = (float*)   (ws + alloc_off(cur, (size_t)R_CNT * NH * F_OUT * 4));
  _Float16* h1T    = (_Float16*)(ws + alloc_off(cur, (size_t)R_CNT * NH * F_OUT * 2));
  _Float16* attF   = (_Float16*)(ws + alloc_off(cur, (size_t)S_CNT * R_CNT * 2));
  _Float16* attT   = (_Float16*)(ws + alloc_off(cur, (size_t)S_CNT * R_CNT * 2));
  float*    ubuf   = (float*)   (ws + alloc_off(cur, (size_t)S_CNT * F_OUT * 4));
  _Float16* uf16   = (_Float16*)(ws + alloc_off(cur, (size_t)S_CNT * F_OUT * 2));
  float*    vbuf   = (float*)   (ws + alloc_off(cur, (size_t)R_CNT * F_OUT * 4));
  _Float16* vf16   = (_Float16*)(ws + alloc_off(cur, (size_t)R_CNT * F_OUT * 2));
  _Float16* xf16   = (_Float16*)(ws + alloc_off(cur, (size_t)S_CNT * NH * R_CNT * 2));
  float*    hbuf   = (float*)   (ws + alloc_off(cur, (size_t)S_CNT * R_CNT * 4));
  float*    s2     = (float*)   (ws + alloc_off(cur, (size_t)S_CNT * 4));
  float*    r1     = (float*)   (ws + alloc_off(cur, (size_t)R_CNT * 4));
  float*    avb    = (float*)   (ws + alloc_off(cur, (size_t)R_CNT * 4));
  float*    ebuf   = (float*)   (ws + alloc_off(cur, (size_t)S_CNT * 4));
  float*    mu     = (float*)   (ws + alloc_off(cur, (size_t)F_OUT * 4));
  float*    var    = (float*)   (ws + alloc_off(cur, (size_t)F_OUT * 4));

  // ---- precision staging ----
  cvt_f16_kernel<<<4096, 256, 0, stream>>>(Sfeat, Sf16, (size_t)S_CNT * F_IN);
  cvt_f16_kernel<<<1024, 256, 0, stream>>>(Rfeat, Rf16, (size_t)R_CNT * F_IN);
  cvt_catT_kernel<<<1024, 256, 0, stream>>>(W1, W1cT);
  cvt_catT_kernel<<<1024, 256, 0, stream>>>(W2, W2cT);
  // W_out (4096x1024) -> WoT (1024x4096) f16
  transpose_cvt_kernel<<<dim3(R_CNT / 32, (NH * R_CNT) / 32), 256, 0, stream>>>(
      W_out, WoT, NH * R_CNT, R_CNT);
  av_kernel<<<R_CNT / 256, 256, 0, stream>>>(a_out, avb);

  // ---- h2 = S @ [W2_h], h1 = R @ [W1_h] (head-fused, B^T = WcT) ----
  gemm_wmma<0><<<dim3((NH * F_OUT) / 128, S_CNT / 128), 256, 0, stream>>>(
      Sf16, F_IN, W2cT, F_IN, h2, NH * F_OUT, nullptr, 0, S_CNT, NH * F_OUT, F_IN);
  gemm_wmma<0><<<dim3((NH * F_OUT) / 128, R_CNT / 128), 256, 0, stream>>>(
      Rf16, F_IN, W1cT, F_IN, h1, NH * F_OUT, nullptr, 0, R_CNT, NH * F_OUT, F_IN);
  // transposed f16 copies for the B^T sides of att GEMMs
  transpose_cvt_kernel<<<dim3((NH * F_OUT) / 32, S_CNT / 32), 256, 0, stream>>>(
      h2, h2T, S_CNT, NH * F_OUT);
  transpose_cvt_kernel<<<dim3((NH * F_OUT) / 32, R_CNT / 32), 256, 0, stream>>>(
      h1, h1T, R_CNT, NH * F_OUT);

  // ---- per-head attention ----
  for (int h = 0; h < NH; ++h) {
    const float* a1 = a_att + (size_t)h * 2 * F_OUT;
    const float* a2 = a_att + (size_t)h * 2 * F_OUT + F_OUT;
    dot256_kernel<<<S_CNT, 256, 0, stream>>>(h2, NH * F_OUT, h * F_OUT, a2, s2);
    dot256_kernel<<<R_CNT, 256, 0, stream>>>(h1, NH * F_OUT, h * F_OUT, a1, r1);
    att_kernel<<<S_CNT, 256, 0, stream>>>(s2, r1, adj, attF, attT);

    // u_pre = att @ h1[:,h]    (16384 x 256 x 1024), B^T = h1T rows [h*256, h*256+256)
    gemm_wmma<0><<<dim3(F_OUT / 128, S_CNT / 128), 256, 0, stream>>>(
        attF, R_CNT, h1T + (size_t)h * F_OUT * R_CNT, R_CNT,
        ubuf, F_OUT, nullptr, 0, S_CNT, F_OUT, R_CNT);
    // v_pre = att^T @ h2[:,h]  (1024 x 256 x 16384), B^T = h2T rows [h*256, ...)
    gemm_wmma<0><<<dim3(F_OUT / 128, R_CNT / 128), 256, 0, stream>>>(
        attT, S_CNT, h2T + (size_t)h * F_OUT * S_CNT, S_CNT,
        vbuf, F_OUT, nullptr, 0, R_CNT, F_OUT, S_CNT);

    // BN + leaky_relu (reference: bn2 on u, bn1 on v)
    bn_stats_kernel<<<F_OUT, 256, 0, stream>>>(ubuf, S_CNT, mu, var);
    bn_apply_kernel<<<4096, 256, 0, stream>>>(ubuf, (size_t)S_CNT * F_OUT,
        mu, var, bn2_g + (size_t)h * F_OUT, bn2_b + (size_t)h * F_OUT, uf16);
    bn_stats_kernel<<<F_OUT, 256, 0, stream>>>(vbuf, R_CNT, mu, var);
    bn_apply_kernel<<<1024, 256, 0, stream>>>(vbuf, (size_t)R_CNT * F_OUT,
        mu, var, bn1_g + (size_t)h * F_OUT, bn1_b + (size_t)h * F_OUT, vf16);

    // head_out = elu(u @ v^T) -> f16 slice of x (col offset h*R_CNT, row stride NH*R_CNT)
    gemm_wmma<1><<<dim3(R_CNT / 128, S_CNT / 128), 256, 0, stream>>>(
        uf16, F_OUT, vf16, F_OUT,
        nullptr, 0, xf16 + (size_t)h * R_CNT, NH * R_CNT, S_CNT, R_CNT, F_OUT);
  }

  // ---- output attention: h = x @ W_out (16384 x 1024 x 4096), B^T = WoT ----
  gemm_wmma<0><<<dim3(R_CNT / 128, S_CNT / 128), 256, 0, stream>>>(
      xf16, NH * R_CNT, WoT, NH * R_CNT, hbuf, R_CNT, nullptr, 0,
      S_CNT, R_CNT, NH * R_CNT);

  eout_kernel<<<S_CNT, 256, 0, stream>>>(hbuf, avb, ebuf);
  final_kernel<<<S_CNT, 256, 0, stream>>>(hbuf, ebuf, adj, out);
}
